// DEQModel_11690900980030
// MI455X (gfx1250) — matrix operand order repaired and checked by reading, not compiled, not run on hardware
//
#include <hip/hip_runtime.h>
#include <hip/hip_bf16.h>

typedef __attribute__((ext_vector_type(16))) _Float16 v16h;
typedef __attribute__((ext_vector_type(8)))  float    v8f;

#define B_   8
#define H_   128
#define W_   200
#define C48  48
#define C64  64
#define NGRP 8
#define EPSF 1e-5f
#define HW   (H_ * W_)
#define N48  (B_ * C48 * HW)   /* 9,830,400  */
#define N64  (B_ * C64 * HW)   /* 13,107,200 */
#define K1   (C48 * 9)         /* 432 */
#define K2   (C64 * 9)         /* 576 */
#define KT1  14                /* ceil(432/32) */
#define KT2  18                /* 576/32 */
#define WTIL 13                /* ceil(200/16) */
#define CP1  56                /* LDS channel stride conv1 (48+8) */
#define CP2  72                /* LDS channel stride conv2 (64+8) */

// ---------------------------------------------------------------------------
// Pack conv weights [Cout][Cin][3][3] into WMMA A-fragment layout (f16).
// K order: k = (kh*3+kw)*Cin + c   (channel-innermost runs of Cin).
// A 16x32 tile per (mt,kt): lane l<16 holds M=mt*16+l, K {kt*32+(0..15)};
// lane l>=16 holds same M, K {16..31}.  packed[((mt*KT+kt)*32+lane)*16+e]
// ---------------------------------------------------------------------------
__global__ void k_prepack(const float* __restrict__ w, _Float16* __restrict__ packed,
                          int Cout, int Cin, int KT) {
    int Ktot  = Cin * 9;
    int total = (Cout / 16) * KT * 32 * 16;
    for (int t = blockIdx.x * blockDim.x + threadIdx.x; t < total;
         t += gridDim.x * blockDim.x) {
        int e    = t & 15;
        int lane = (t >> 4) & 31;
        int kt   = (t >> 9) % KT;
        int mt   = t / (512 * KT);
        int base = (lane < 16) ? 0 : 16;
        int k    = kt * 32 + base + e;
        int m    = mt * 16 + (lane & 15);
        float v  = 0.0f;
        if (k < Ktot) {
            int run = k / Cin;          // kh*3+kw
            int c   = k - run * Cin;
            v = w[(m * Cin + c) * 9 + run];
        }
        packed[t] = (_Float16)v;
    }
}

// ---------------------------------------------------------------------------
// conv0: 1->48, 3x3, pad 1, + bias   (tiny, direct)
// ---------------------------------------------------------------------------
__global__ void k_conv0(const float* __restrict__ x, const float* __restrict__ w,
                        const float* __restrict__ bias, float* __restrict__ out) {
    int i = blockIdx.x * blockDim.x + threadIdx.x;
    if (i >= N48) return;
    int wx = i % W_;  int t = i / W_;
    int hy = t % H_;  t /= H_;
    int c  = t % C48; int b = t / C48;
    float acc = bias[c];
    const float* xin = x + b * HW;
    const float* wc  = w + c * 9;
    #pragma unroll
    for (int kh = 0; kh < 3; kh++) {
        int gy = hy + kh - 1;
        if ((unsigned)gy >= (unsigned)H_) continue;
        #pragma unroll
        for (int kw = 0; kw < 3; kw++) {
            int gx = wx + kw - 1;
            if ((unsigned)gx >= (unsigned)W_) continue;
            acc += xin[gy * W_ + gx] * wc[kh * 3 + kw];
        }
    }
    out[i] = acc;
}

// ---------------------------------------------------------------------------
// BatchNorm stats: one block per channel (48), reduce over B*H*W
// ---------------------------------------------------------------------------
__global__ void k_bn_stats(const float* __restrict__ t, float* __restrict__ mu,
                           float* __restrict__ rs) {
    int c = blockIdx.x;
    const int count = B_ * HW;
    float s = 0.f, s2 = 0.f;
    for (int i = threadIdx.x; i < count; i += blockDim.x) {
        int b = i / HW, r = i % HW;
        float v = t[(b * C48 + c) * HW + r];
        s += v; s2 += v * v;
    }
    __shared__ float r1[256], r2[256];
    r1[threadIdx.x] = s; r2[threadIdx.x] = s2;
    __syncthreads();
    for (int o = 128; o > 0; o >>= 1) {
        if ((int)threadIdx.x < o) { r1[threadIdx.x] += r1[threadIdx.x + o];
                                    r2[threadIdx.x] += r2[threadIdx.x + o]; }
        __syncthreads();
    }
    if (threadIdx.x == 0) {
        float m = r1[0] / count;
        float v = r2[0] / count - m * m;
        mu[c] = m; rs[c] = rsqrtf(v + EPSF);
    }
}

__global__ void k_bn_apply(const float* __restrict__ t, const float* __restrict__ mu,
                           const float* __restrict__ rs, const float* __restrict__ g,
                           const float* __restrict__ bb, float* __restrict__ h) {
    int i = blockIdx.x * blockDim.x + threadIdx.x;
    if (i >= N48) return;
    int c = (i / HW) % C48;
    h[i] = (t[i] - mu[c]) * rs[c] * g[c] + bb[c];
}

// ---------------------------------------------------------------------------
// GroupNorm stats: one block per (b, group); C = 64 or 48
// ---------------------------------------------------------------------------
__global__ void k_gn_stats(const float* __restrict__ x, float* __restrict__ mu,
                           float* __restrict__ rs, int C) {
    int bg = blockIdx.x;
    int b = bg / NGRP, g = bg % NGRP;
    int Cg = C / NGRP;
    int count = Cg * HW;
    const float* base = x + (size_t)(b * C + g * Cg) * HW;
    float s = 0.f, s2 = 0.f;
    for (int i = threadIdx.x; i < count; i += blockDim.x) {
        float v = base[i];
        s += v; s2 += v * v;
    }
    __shared__ float r1[256], r2[256];
    r1[threadIdx.x] = s; r2[threadIdx.x] = s2;
    __syncthreads();
    for (int o = 128; o > 0; o >>= 1) {
        if ((int)threadIdx.x < o) { r1[threadIdx.x] += r1[threadIdx.x + o];
                                    r2[threadIdx.x] += r2[threadIdx.x + o]; }
        __syncthreads();
    }
    if (threadIdx.x == 0) {
        float m = r1[0] / count;
        float v = r2[0] / count - m * m;
        mu[bg] = m; rs[bg] = rsqrtf(v + EPSF);
    }
}

// out[i] = (in[i]-mu)*rs*g[c]+b[c]   (gn3 -> F slot)
__global__ void k_gn_apply(const float* __restrict__ in, float* __restrict__ out,
                           const float* __restrict__ mu, const float* __restrict__ rs,
                           const float* __restrict__ g, const float* __restrict__ bb,
                           int C, int total) {
    int i = blockIdx.x * blockDim.x + threadIdx.x;
    if (i >= total) return;
    int c = (i / HW) % C;
    int b = i / (HW * C);
    int grp = c / (C / NGRP);
    float m = mu[b * NGRP + grp], r = rs[b * NGRP + grp];
    out[i] = (in[i] - m) * r * g[c] + bb[c];
}

// t[i] = relu( z[i] + gn2(t[i]) )    (in-place, C=48)
__global__ void k_gn2_addrelu(float* __restrict__ t, const float* __restrict__ z,
                              const float* __restrict__ mu, const float* __restrict__ rs,
                              const float* __restrict__ g, const float* __restrict__ bb) {
    int i = blockIdx.x * blockDim.x + threadIdx.x;
    if (i >= N48) return;
    int c = (i / HW) % C48;
    int b = i / (HW * C48);
    int grp = c / (C48 / NGRP);
    float v = (t[i] - mu[b * NGRP + grp]) * rs[b * NGRP + grp] * g[c] + bb[c];
    v += z[i];
    t[i] = v > 0.f ? v : 0.f;
}

// ---------------------------------------------------------------------------
// conv1 (48->64) implicit GEMM via v_wmma_f32_16x16x32_f16, + ReLU epilogue.
// Patch LDS layout: [kh][col][c], channel stride CP1; B-fragment = one
// contiguous v16h LDS read (k = run*48 + c ordering; 16-reads never cross run).
// Grid (WTIL, H, B), 128 threads (4 waves; wave=mt -> cout 16*mt..+15).
// ---------------------------------------------------------------------------
__global__ __launch_bounds__(128)
void k_conv1(const float* __restrict__ z, const _Float16* __restrict__ packedA,
             float* __restrict__ y) {
    __shared__ __align__(32) _Float16 patch[3 * 18 * CP1];
    int w0 = blockIdx.x * 16;
    int hy = blockIdx.y;
    int b  = blockIdx.z;
    int tid = threadIdx.x;
    const float* zb = z + (size_t)b * C48 * HW;
    for (int i = tid; i < C48 * 3 * 18; i += 128) {
        int col = i % 18; int t = i / 18;
        int kh = t % 3;   int c = t / 3;
        int gy = hy + kh - 1;
        int gx = w0 + col - 1;
        float v = 0.f;
        if ((unsigned)gy < (unsigned)H_ && (unsigned)gx < (unsigned)W_)
            v = zb[(c * H_ + gy) * W_ + gx];
        patch[(kh * 18 + col) * CP1 + c] = (_Float16)v;
    }
    __syncthreads();

    int lane  = tid & 31;
    int mt    = tid >> 5;
    int col   = lane & 15;
    int koff0 = (lane & 16) ? 16 : 0;
    v8f acc = {};
    const v16h* A = (const v16h*)packedA + (size_t)mt * KT1 * 32 + lane;
    #pragma unroll
    for (int kt = 0; kt < KT1; kt++) {
        v16h a = A[(size_t)kt * 32];
        int base = kt * 32 + koff0;
        if (base > K1 - 16) base = K1 - 16;   // tail: A is zero there anyway
        int run = base / C48;                  // kh*3+kw, 0..8
        int c0  = base - run * C48;            // multiple of 16
        int kh  = run / 3;
        int kw  = run - kh * 3;
        v16h bf = *(const v16h*)(patch + (kh * 18 + col + kw) * CP1 + c0);
        acc = __builtin_amdgcn_wmma_f32_16x16x32_f16(false, a, false, bf,
                                                     (short)0, acc, false, false);
    }
    int wgx = w0 + col;
    if (wgx < W_) {
        int mbase = mt * 16 + ((lane & 16) ? 8 : 0);
        float* yo = y + (size_t)b * C64 * HW + hy * W_ + wgx;
        #pragma unroll
        for (int r = 0; r < 8; r++) {
            float v = acc[r];
            yo[(size_t)(mbase + r) * HW] = v > 0.f ? v : 0.f;
        }
    }
}

// ---------------------------------------------------------------------------
// conv2 (64->48) implicit GEMM via WMMA.
// gn1 (C=64) is FUSED into the patch load (normalize while converting to f16);
// boundary padding stays raw zero, matching the reference order.
// Epilogue: t = acc + h.
// ---------------------------------------------------------------------------
__global__ __launch_bounds__(128)
void k_conv2(const float* __restrict__ y, const _Float16* __restrict__ packedA,
             const float* __restrict__ hbuf, float* __restrict__ tout,
             const float* __restrict__ gmu, const float* __restrict__ grs,
             const float* __restrict__ g1g, const float* __restrict__ g1b) {
    __shared__ __align__(32) _Float16 patch[3 * 18 * CP2];
    int w0 = blockIdx.x * 16;
    int hy = blockIdx.y;
    int b  = blockIdx.z;
    int tid = threadIdx.x;
    const float* yb = y + (size_t)b * C64 * HW;
    for (int i = tid; i < C64 * 3 * 18; i += 128) {
        int col = i % 18; int t = i / 18;
        int kh = t % 3;   int c = t / 3;
        int gy = hy + kh - 1;
        int gx = w0 + col - 1;
        float v = 0.f;
        if ((unsigned)gy < (unsigned)H_ && (unsigned)gx < (unsigned)W_) {
            int grp = c >> 3;                       // C64 / NGRP = 8
            float raw = yb[(c * H_ + gy) * W_ + gx];
            v = (raw - gmu[b * NGRP + grp]) * grs[b * NGRP + grp] * g1g[c] + g1b[c];
        }
        patch[(kh * 18 + col) * CP2 + c] = (_Float16)v;
    }
    __syncthreads();

    int lane = tid & 31;
    int mt   = tid >> 5;
    if (mt >= 3) return;                     // 3 M-tiles (48 channels)
    int col   = lane & 15;
    int koff0 = (lane & 16) ? 16 : 0;
    v8f acc = {};
    const v16h* A = (const v16h*)packedA + (size_t)mt * KT2 * 32 + lane;
    #pragma unroll
    for (int kt = 0; kt < KT2; kt++) {
        v16h a = A[(size_t)kt * 32];
        int base = kt * 32 + koff0;          // K2 = 576 exact, no clamp
        int run = base >> 6;                 // /64
        int c0  = base & 63;                 // multiple of 16
        int kh  = run / 3;
        int kw  = run - kh * 3;
        v16h bf = *(const v16h*)(patch + (kh * 18 + col + kw) * CP2 + c0);
        acc = __builtin_amdgcn_wmma_f32_16x16x32_f16(false, a, false, bf,
                                                     (short)0, acc, false, false);
    }
    int wgx = w0 + col;
    if (wgx < W_) {
        int mbase = mt * 16 + ((lane & 16) ? 8 : 0);
        size_t base2 = (size_t)b * C48 * HW + hy * W_ + wgx;
        #pragma unroll
        for (int r = 0; r < 8; r++) {
            size_t idx = base2 + (size_t)(mbase + r) * HW;
            tout[idx] = acc[r] + hbuf[idx];
        }
    }
}

// ---------------------------------------------------------------------------
// Anderson: Gram partials  G_i . G_j  (G = F - X), per batch.
// ---------------------------------------------------------------------------
__global__ void k_gram_partial(const float* __restrict__ Xs, const float* __restrict__ Fs,
                               float* __restrict__ partial, int n) {
    int chunk = blockIdx.x;
    int b     = blockIdx.y;
    int npairs = n * (n + 1) / 2;
    const long D = (long)C48 * HW;
    float p[15];
    #pragma unroll
    for (int q = 0; q < 15; q++) p[q] = 0.f;
    for (long d = (long)chunk * 256 + threadIdx.x; d < D; d += 64L * 256) {
        float gv[5];
        for (int i = 0; i < n; i++) {
            size_t idx = (size_t)i * N48 + (size_t)b * D + d;
            gv[i] = Fs[idx] - Xs[idx];
        }
        int q = 0;
        for (int i = 0; i < n; i++)
            for (int j = i; j < n; j++)
                p[q++] += gv[i] * gv[j];
    }
    __shared__ float red[256];
    for (int q = 0; q < npairs; q++) {
        red[threadIdx.x] = p[q];
        __syncthreads();
        for (int o = 128; o > 0; o >>= 1) {
            if ((int)threadIdx.x < o) red[threadIdx.x] += red[threadIdx.x + o];
            __syncthreads();
        }
        if (threadIdx.x == 0) partial[(b * 64 + chunk) * 15 + q] = red[0];
        __syncthreads();
    }
}

// Reduce partials, build bordered system Hb, solve (Gaussian elim, 1 thr/batch)
__global__ void k_gram_solve(const float* __restrict__ partial,
                             float* __restrict__ alpha, int n) {
    int b = blockIdx.x;
    __shared__ float pair[15];
    int npairs = n * (n + 1) / 2;
    if ((int)threadIdx.x < npairs) {
        float s = 0.f;
        for (int c = 0; c < 64; c++) s += partial[(b * 64 + c) * 15 + threadIdx.x];
        pair[threadIdx.x] = s;
    }
    __syncthreads();
    if (threadIdx.x == 0) {
        float Hm[5][5];
        int q = 0;
        for (int i = 0; i < n; i++)
            for (int j = i; j < n; j++) { Hm[i][j] = pair[q]; Hm[j][i] = pair[q]; q++; }
        for (int i = 0; i < n; i++) Hm[i][i] += 1e-4f;
        int m = n + 1;
        float A[6][7];
        for (int i = 0; i < m; i++)
            for (int j = 0; j < m; j++)
                A[i][j] = (i == 0 && j == 0) ? 0.f
                        : (i == 0 || j == 0) ? 1.f
                        : Hm[i - 1][j - 1];
        for (int i = 0; i < m; i++) A[i][m] = (i == 0) ? 1.f : 0.f;
        for (int cpos = 0; cpos < m; cpos++) {
            int piv = cpos; float best = fabsf(A[cpos][cpos]);
            for (int r = cpos + 1; r < m; r++) {
                float v = fabsf(A[r][cpos]);
                if (v > best) { best = v; piv = r; }
            }
            if (piv != cpos)
                for (int j = cpos; j <= m; j++) {
                    float tt = A[cpos][j]; A[cpos][j] = A[piv][j]; A[piv][j] = tt;
                }
            float inv = 1.f / A[cpos][cpos];
            for (int j = cpos; j <= m; j++) A[cpos][j] *= inv;
            for (int r = 0; r < m; r++)
                if (r != cpos) {
                    float f = A[r][cpos];
                    if (f != 0.f)
                        for (int j = cpos; j <= m; j++) A[r][j] -= f * A[cpos][j];
                }
        }
        for (int j = 0; j < n; j++) alpha[b * 5 + j] = A[j + 1][m];
    }
}

// xk = sum_j alpha[b][j] * F_j   (beta == 1.0)
__global__ void k_combine(const float* __restrict__ Fs, const float* __restrict__ alpha,
                          float* __restrict__ xout, int n) {
    int i = blockIdx.x * blockDim.x + threadIdx.x;
    if (i >= N48) return;
    int b = i / (C48 * HW);
    float s = 0.f;
    for (int j = 0; j < n; j++) s += alpha[b * 5 + j] * Fs[(size_t)j * N48 + i];
    xout[i] = s;
}

// out[b,c,h,o] = sum_w z[b,c,h,w] * fc_w[o,w] + fc_b[o]
__global__ void k_fc(const float* __restrict__ z, const float* __restrict__ fw,
                     const float* __restrict__ fb, float* __restrict__ out) {
    int i = blockIdx.x * blockDim.x + threadIdx.x;
    const int total = B_ * C48 * H_ * 10;
    if (i >= total) return;
    int o   = i % 10;
    int row = i / 10;
    const float* zr = z + (size_t)row * W_;
    const float* wr = fw + o * W_;
    float s = fb[o];
    for (int w = 0; w < W_; w++) s += zr[w] * wr[w];
    out[i] = s;
}

// ---------------------------------------------------------------------------
extern "C" void kernel_launch(void* const* d_in, const int* in_sizes, int n_in,
                              void* d_out, int out_size, void* d_ws, size_t ws_size,
                              hipStream_t stream) {
    (void)in_sizes; (void)n_in; (void)out_size; (void)ws_size;
    const float* x       = (const float*)d_in[0];
    const float* conv0_w = (const float*)d_in[1];
    const float* conv0_b = (const float*)d_in[2];
    const float* bn_g    = (const float*)d_in[3];
    const float* bn_b    = (const float*)d_in[4];
    const float* conv1_w = (const float*)d_in[5];
    const float* conv2_w = (const float*)d_in[6];
    const float* gn1_g   = (const float*)d_in[7];
    const float* gn1_b   = (const float*)d_in[8];
    const float* gn2_g   = (const float*)d_in[9];
    const float* gn2_b   = (const float*)d_in[10];
    const float* gn3_g   = (const float*)d_in[11];
    const float* gn3_b   = (const float*)d_in[12];
    const float* fc_w    = (const float*)d_in[13];
    const float* fc_b    = (const float*)d_in[14];
    float* out = (float*)d_out;

    char* ws = (char*)d_ws;
    size_t off = 0;
    auto alloc = [&](size_t bytes) -> void* {
        void* p = ws + off;
        off += (bytes + 255) & ~(size_t)255;
        return p;
    };
    float*    T   = (float*)alloc((size_t)N48 * 4);      // conv0 out / t / u
    float*    Hh  = (float*)alloc((size_t)N48 * 4);      // h (input injection)
    float*    Y   = (float*)alloc((size_t)N64 * 4);      // conv1 output (pre-GN)
    float*    Xs  = (float*)alloc((size_t)5 * N48 * 4);  // Anderson X history
    float*    Fs  = (float*)alloc((size_t)5 * N48 * 4);  // Anderson F history
    _Float16* pA1 = (_Float16*)alloc((size_t)4 * KT1 * 32 * 16 * 2);
    _Float16* pA2 = (_Float16*)alloc((size_t)3 * KT2 * 32 * 16 * 2);
    float* bmu = (float*)alloc(48 * 4);
    float* brs = (float*)alloc(48 * 4);
    float* gmu = (float*)alloc(64 * 4);
    float* grs = (float*)alloc(64 * 4);
    float* part = (float*)alloc((size_t)8 * 64 * 15 * 4);
    float* alp  = (float*)alloc(40 * 4);

    const int T256_48 = (N48 + 255) / 256;
    dim3 cgrid(WTIL, H_, B_);

    // Pre-pack WMMA weight fragments
    k_prepack<<<64, 256, 0, stream>>>(conv1_w, pA1, C64, C48, KT1);
    k_prepack<<<64, 256, 0, stream>>>(conv2_w, pA2, C48, C64, KT2);

    // h = BN(conv0(x) + b)
    k_conv0<<<T256_48, 256, 0, stream>>>(x, conv0_w, conv0_b, T);
    k_bn_stats<<<48, 256, 0, stream>>>(T, bmu, brs);
    k_bn_apply<<<T256_48, 256, 0, stream>>>(T, bmu, brs, bn_g, bn_b, Hh);

    // X0 = 0
    hipMemsetAsync(Xs, 0, (size_t)N48 * 4, stream);

    auto eval_f = [&](const float* zin, float* fout) {
        k_conv1<<<cgrid, 128, 0, stream>>>(zin, pA1, Y);
        k_gn_stats<<<64, 256, 0, stream>>>(Y, gmu, grs, C64);
        k_conv2<<<cgrid, 128, 0, stream>>>(Y, pA2, Hh, T, gmu, grs, gn1_g, gn1_b);
        k_gn_stats<<<64, 256, 0, stream>>>(T, gmu, grs, C48);
        k_gn2_addrelu<<<T256_48, 256, 0, stream>>>(T, zin, gmu, grs, gn2_g, gn2_b);
        k_gn_stats<<<64, 256, 0, stream>>>(T, gmu, grs, C48);
        k_gn_apply<<<T256_48, 256, 0, stream>>>(T, fout, gmu, grs, gn3_g, gn3_b, C48, N48);
    };

    // F0 = f(X0); X1 = F0; F1 = f(X1)
    eval_f(Xs, Fs);
    hipMemcpyAsync(Xs + (size_t)N48, Fs, (size_t)N48 * 4,
                   hipMemcpyDeviceToDevice, stream);
    eval_f(Xs + (size_t)N48, Fs + (size_t)N48);

    // Anderson iterations k = 2..24
    for (int k = 2; k < 25; k++) {
        int n    = (k < 5) ? k : 5;
        int slot = k % 5;
        k_gram_partial<<<dim3(64, 8), 256, 0, stream>>>(Xs, Fs, part, n);
        k_gram_solve<<<8, 64, 0, stream>>>(part, alp, n);
        k_combine<<<T256_48, 256, 0, stream>>>(Fs, alp, Xs + (size_t)slot * N48, n);
        eval_f(Xs + (size_t)slot * N48, Fs + (size_t)slot * N48);
    }

    // Final z = f(z_star) == F slot 4; then Linear(200,10)
    const float* zfin = Fs + (size_t)4 * N48;
    int fct = (B_ * C48 * H_ * 10 + 255) / 256;
    k_fc<<<fct, 256, 0, stream>>>(zfin, fc_w, fc_b, out);
}
